// _MultiheadNonlocalNd_14654428414739
// MI455X (gfx1250) — matrix operand order repaired and checked by reading, not compiled, not run on hardware
//
#include <hip/hip_runtime.h>
#include <hip/hip_bf16.h>
#include <math.h>

// ---------------------------------------------------------------------------
// Multi-head non-local attention (B=2, C=256, H=W=64 -> N=4096, heads=4, D=64)
// Fused flash-attention in f16 WMMA (v_wmma_f32_16x16x32_f16), f32 accumulate.
// K/V staged with gfx1250 async copies (global_load_async_to_lds_b128,
// ASYNCcnt) into double-buffered LDS; each wave owns TWO 16-row query strips
// so every K/V B-fragment feeds two WMMAs (2x arithmetic intensity vs LDS).
// Workspace layout (assumes ws_size >= 16MB):
//   q : [B*NH, N, D] f16   (pre-scaled by D^-0.5)   offset 0
//   k : [B*NH, N, D] f16                            offset 1*QKV
//   v : [B*NH, N, D] f16                            offset 2*QKV
//   y : [B*NH, N, D] f16   (attention output)       offset 3*QKV
// ---------------------------------------------------------------------------

typedef __attribute__((ext_vector_type(16))) _Float16 v16h;
typedef __attribute__((ext_vector_type(8)))  float    v8f;
typedef __attribute__((ext_vector_type(4)))  _Float16 h4;

#define CDIM 256
#define NH   4
#define DDIM 64
#define NTOK 4096
#define QKV_ELEMS ((size_t)2 * NH * NTOK * DDIM)   // 2,097,152 halfs per tensor

// A-matrix (16x32 f16) per-lane K index: lane group g = lane>>4, row M = lane&15
__device__ __forceinline__ int a_kidx(int i, int g) {
    return i + 16 * g;
}

// gfx1250 async copy: 16B global -> LDS, tracked by ASYNCcnt.
// vdst = LDS byte address (low 32 bits of the generic __shared__ pointer).
__device__ __forceinline__ void async_ld_b128(void* lds_ptr, const void* gptr) {
    uint32_t loff = (uint32_t)(uintptr_t)lds_ptr;
    asm volatile("global_load_async_to_lds_b128 %0, %1, off"
                 :: "v"(loff), "v"(gptr)
                 : "memory");
}
#define S_WAIT_ASYNC(n) asm volatile("s_wait_asynccnt %0" :: "i"(n) : "memory")

// ---------------------------------------------------------------------------
// Kernel 1: 1x1 embed conv -> q (scaled, f16), k (f16), v (f16) in [bh, n, d]
// grid (N/64, 192/4, B*NH), block 256 = (64 n) x (4 d)
// ---------------------------------------------------------------------------
__global__ __launch_bounds__(256) void embed_kernel(
    const float* __restrict__ x, const float* __restrict__ w,
    const float* __restrict__ bias,
    _Float16* __restrict__ q, _Float16* __restrict__ k, _Float16* __restrict__ v)
{
    const int n  = blockIdx.x * 64 + (threadIdx.x & 63);
    const int d  = blockIdx.y * 4 + (threadIdx.x >> 6);
    const int bh = blockIdx.z;
    const int b  = bh >> 2;
    const int h  = bh & 3;

    const float* xp = x + (size_t)b * CDIM * NTOK + n;
    const float* wp = w + ((size_t)h * 192 + d) * CDIM;

    float acc = bias[h * 192 + d];
#pragma unroll 4
    for (int c = 0; c < CDIM; ++c)
        acc += xp[(size_t)c * NTOK] * wp[c];

    const size_t ob = ((size_t)bh * NTOK + n) * DDIM;
    if (d < 64)        q[ob + d]        = (_Float16)(acc * 0.125f);   // D^-0.5
    else if (d < 128)  k[ob + (d - 64)] = (_Float16)acc;
    else               v[ob + (d - 128)] = (_Float16)acc;
}

// ---------------------------------------------------------------------------
// Kernel 2: flash attention.  grid (N/128, B*NH), block 128 (4 waves).
// Each wave owns two 16-row query strips (rows wave*16 and 64+wave*16 of the
// work-group's 128-row block); iterates over 64 key blocks of 64 keys,
// K/V double-buffered via async LDS copies.
// ---------------------------------------------------------------------------
__global__ __launch_bounds__(128) void attn_kernel(
    const _Float16* __restrict__ q, const _Float16* __restrict__ k,
    const _Float16* __restrict__ v, _Float16* __restrict__ y)
{
    const int nblk = blockIdx.x;
    const int bh   = blockIdx.y;
    const int tid  = threadIdx.x;
    const int lane = tid & 31;
    const int wave = tid >> 5;
    const int lm   = lane & 15;
    const int g    = lane >> 4;

    __shared__ _Float16 Qs[128][72];
    __shared__ _Float16 Ks[2][64][72];
    __shared__ _Float16 Vs[2][64][72];
    __shared__ _Float16 Ps[4][32][72];   // per wave: 2 strips x 16 rows

    const _Float16* qbase = q + ((size_t)bh * NTOK + (size_t)nblk * 128) * DDIM;
    const _Float16* kbase = k + (size_t)bh * NTOK * DDIM;
    const _Float16* vbase = v + (size_t)bh * NTOK * DDIM;

    // async copies for a key-block (8 x b128 per thread: 4 K + 4 V)
    auto issue_kv = [&](int kb, int buf) {
        const _Float16* kp = kbase + (size_t)kb * 64 * DDIM;
        const _Float16* vp = vbase + (size_t)kb * 64 * DDIM;
#pragma unroll
        for (int it = 0; it < 4; ++it) {
            int idx = it * 128 + tid;          // 512 chunks of 8 halfs / tensor
            int r = idx >> 3, c = (idx & 7) * 8;
            async_ld_b128(&Ks[buf][r][c], kp + (size_t)r * DDIM + c);
            async_ld_b128(&Vs[buf][r][c], vp + (size_t)r * DDIM + c);
        }
    };
    issue_kv(0, 0);

    // stage the 128x64 Q block (regular loads)
#pragma unroll
    for (int it = 0; it < 8; ++it) {
        int idx = it * 128 + tid;              // 1024 chunks of 8 halfs
        int r = idx >> 3, c = (idx & 7) * 8;
        *(h4*)&Qs[r][c]     = *(const h4*)&qbase[(size_t)r * DDIM + c];
        *(h4*)&Qs[r][c + 4] = *(const h4*)&qbase[(size_t)r * DDIM + c + 4];
    }
    __syncthreads();

    // Q fragments (A layout) for both strips
    v16h aq[2][2];                             // [strip][k-step]
#pragma unroll
    for (int st = 0; st < 2; ++st) {
        const int m = st * 64 + wave * 16 + lm;
#pragma unroll
        for (int i = 0; i < 16; ++i) {
            int kk = a_kidx(i, g);
            aq[st][0][i] = Qs[m][kk];
            aq[st][1][i] = Qs[m][32 + kk];
        }
    }

    float mrow[2][8], lrow[2][8];
    v8f o[2][4];
#pragma unroll
    for (int st = 0; st < 2; ++st) {
#pragma unroll
        for (int r = 0; r < 8; ++r) { mrow[st][r] = -1e30f; lrow[st][r] = 0.0f; }
#pragma unroll
        for (int dt = 0; dt < 4; ++dt)
#pragma unroll
            for (int r = 0; r < 8; ++r) o[st][dt][r] = 0.0f;
    }

    for (int kb = 0; kb < NTOK / 64; ++kb) {
        const int buf = kb & 1;
        __syncthreads();   // all waves done with buf^1 before overwriting it
        if (kb + 1 < NTOK / 64) {
            issue_kv(kb + 1, buf ^ 1);
            S_WAIT_ASYNC(8);      // the previous batch (this kb) has landed
        } else {
            S_WAIT_ASYNC(0);
        }
        __syncthreads();   // buf visible work-group wide

        // ---- S = Q K^T: 4 col tiles x 2 strips; bk shared across strips ----
        v8f s[2][4];
#pragma unroll
        for (int nt = 0; nt < 4; ++nt) {
            v16h bk0, bk1;
#pragma unroll
            for (int i = 0; i < 16; ++i) {
                int kk = i + 16 * g;                 // B layout K index
                bk0[i] = Ks[buf][nt * 16 + lm][kk];
                bk1[i] = Ks[buf][nt * 16 + lm][32 + kk];
            }
#pragma unroll
            for (int st = 0; st < 2; ++st) {
                v8f acc;
#pragma unroll
                for (int r = 0; r < 8; ++r) acc[r] = 0.0f;
                acc = __builtin_amdgcn_wmma_f32_16x16x32_f16(false, aq[st][0], false, bk0,
                                                             (short)0, acc, false, false);
                acc = __builtin_amdgcn_wmma_f32_16x16x32_f16(false, aq[st][1], false, bk1,
                                                             (short)0, acc, false, false);
                s[st][nt] = acc;
            }
        }

        // ---- online softmax per strip (rows spread across half-waves) ----
#pragma unroll
        for (int st = 0; st < 2; ++st) {
            float mnew[8];
#pragma unroll
            for (int r = 0; r < 8; ++r) {
                float t = fmaxf(fmaxf(s[st][0][r], s[st][1][r]),
                                fmaxf(s[st][2][r], s[st][3][r]));
                t = fmaxf(t, __shfl_xor(t, 1, 32));
                t = fmaxf(t, __shfl_xor(t, 2, 32));
                t = fmaxf(t, __shfl_xor(t, 4, 32));
                t = fmaxf(t, __shfl_xor(t, 8, 32));
                mnew[r] = fmaxf(mrow[st][r], t);
            }
            float rs[8];
#pragma unroll
            for (int r = 0; r < 8; ++r) {
                float alpha = __expf(mrow[st][r] - mnew[r]);
                lrow[st][r] *= alpha;
#pragma unroll
                for (int dt = 0; dt < 4; ++dt) o[st][dt][r] *= alpha;
                mrow[st][r] = mnew[r];
                rs[r] = 0.0f;
            }
#pragma unroll
            for (int nt = 0; nt < 4; ++nt)
#pragma unroll
                for (int r = 0; r < 8; ++r) {
                    float p = __expf(s[st][nt][r] - mnew[r]);
                    rs[r] += p;
                    Ps[wave][st * 16 + r + 8 * g][nt * 16 + lm] = (_Float16)p;
                }
#pragma unroll
            for (int r = 0; r < 8; ++r) {
                rs[r] += __shfl_xor(rs[r], 1, 32);
                rs[r] += __shfl_xor(rs[r], 2, 32);
                rs[r] += __shfl_xor(rs[r], 4, 32);
                rs[r] += __shfl_xor(rs[r], 8, 32);
                lrow[st][r] += rs[r];
            }
        }

        // ---- O += P V : bv shared across strips ----
        v16h ap[2][2];
#pragma unroll
        for (int st = 0; st < 2; ++st)
#pragma unroll
            for (int i = 0; i < 16; ++i) {
                int kk = a_kidx(i, g);
                ap[st][0][i] = Ps[wave][st * 16 + lm][kk];
                ap[st][1][i] = Ps[wave][st * 16 + lm][32 + kk];
            }
#pragma unroll
        for (int dt = 0; dt < 4; ++dt) {
            v16h bv0, bv1;
#pragma unroll
            for (int i = 0; i < 16; ++i) {
                int kk = i + 16 * g;
                bv0[i] = Vs[buf][kk][dt * 16 + lm];
                bv1[i] = Vs[buf][32 + kk][dt * 16 + lm];
            }
#pragma unroll
            for (int st = 0; st < 2; ++st) {
                o[st][dt] = __builtin_amdgcn_wmma_f32_16x16x32_f16(
                    false, ap[st][0], false, bv0, (short)0, o[st][dt], false, false);
                o[st][dt] = __builtin_amdgcn_wmma_f32_16x16x32_f16(
                    false, ap[st][1], false, bv1, (short)0, o[st][dt], false, false);
            }
        }
    }

    // normalize and store y[bh, n, d] (f16), both strips
#pragma unroll
    for (int st = 0; st < 2; ++st) {
        _Float16* yp = y + ((size_t)bh * NTOK + (size_t)nblk * 128 + st * 64
                            + wave * 16) * DDIM;
#pragma unroll
        for (int r = 0; r < 8; ++r) {
            float invl = 1.0f / lrow[st][r];
#pragma unroll
            for (int dt = 0; dt < 4; ++dt)
                yp[(size_t)(r + 8 * g) * DDIM + dt * 16 + lm]
                    = (_Float16)(o[st][dt][r] * invl);
        }
    }
}

// ---------------------------------------------------------------------------
// Kernel 3: out[b,c,n] = sum_k y[b,k,n] * w_out[k,c] + b_out[c] + x[b,c,n]
// grid (N/64, C/64, B), block 128 (4 waves).  WMMA f16, Y staged async,
// LDS-transposed stores for coalesced [b,c,n] output.
// ---------------------------------------------------------------------------
__global__ __launch_bounds__(128) void proj_kernel(
    const _Float16* __restrict__ y, const float* __restrict__ w_out,
    const float* __restrict__ b_out, const float* __restrict__ x,
    float* __restrict__ out)
{
    const int n0   = blockIdx.x * 64;
    const int c0   = blockIdx.y * 64;
    const int b    = blockIdx.z;
    const int tid  = threadIdx.x;
    const int lane = tid & 31;
    const int wave = tid >> 5;
    const int lm   = lane & 15;
    const int g    = lane >> 4;

    __shared__ _Float16 Ysm[64][40];       // 64 n rows x 32 k
    __shared__ _Float16 Wsm[32][72];       // 32 k rows x 64 c
    __shared__ float    Ts[4][16][68];     // per-wave transpose buffer

    v8f acc[4];
#pragma unroll
    for (int ct = 0; ct < 4; ++ct)
#pragma unroll
        for (int r = 0; r < 8; ++r) acc[ct][r] = 0.0f;

    for (int kb = 0; kb < 8; ++kb) {       // K = 256 in steps of 32
        const int k0   = kb * 32;
        const int head = k0 >> 6;
        const int dl0  = k0 & 63;

        __syncthreads();                   // previous k-step's readers done
        const _Float16* ysrc = y + (((size_t)b * NH + head) * NTOK + n0) * DDIM + dl0;
#pragma unroll
        for (int it = 0; it < 2; ++it) {   // 256 chunks of 8 halfs, async
            int idx = it * 128 + tid;
            int r = idx >> 2, c = (idx & 3) * 8;
            async_ld_b128(&Ysm[r][c], ysrc + (size_t)r * DDIM + c);
        }
#pragma unroll
        for (int it = 0; it < 16; ++it) {  // 2048 f32 -> f16
            int idx = it * 128 + tid;
            int r = idx >> 6, c = idx & 63;
            Wsm[r][c] = (_Float16)w_out[(size_t)(k0 + r) * CDIM + c0 + c];
        }
        S_WAIT_ASYNC(0);
        __syncthreads();

        v16h a;
#pragma unroll
        for (int i = 0; i < 16; ++i)
            a[i] = Ysm[wave * 16 + lm][a_kidx(i, g)];
#pragma unroll
        for (int ct = 0; ct < 4; ++ct) {
            v16h bw;
#pragma unroll
            for (int i = 0; i < 16; ++i)
                bw[i] = Wsm[i + 16 * g][ct * 16 + lm];
            acc[ct] = __builtin_amdgcn_wmma_f32_16x16x32_f16(false, a, false, bw,
                                                             (short)0, acc[ct], false, false);
        }
    }

    // C-layout -> LDS strip [16 n][64 c], then coalesced [c][n] stores
#pragma unroll
    for (int ct = 0; ct < 4; ++ct)
#pragma unroll
        for (int r = 0; r < 8; ++r)
            Ts[wave][r + 8 * g][ct * 16 + lm] = acc[ct][r];

#pragma unroll
    for (int rr = 0; rr < 2; ++rr) {
        int cl = lane * 2 + rr;            // 0..63
        int c  = c0 + cl;
        float bo = b_out[c];
        float*       op = out + ((size_t)b * CDIM + c) * NTOK + n0 + wave * 16;
        const float* xp = x   + ((size_t)b * CDIM + c) * NTOK + n0 + wave * 16;
#pragma unroll
        for (int nn = 0; nn < 16; ++nn)
            op[nn] = Ts[wave][nn][cl] + bo + xp[nn];
    }
}

// ---------------------------------------------------------------------------
extern "C" void kernel_launch(void* const* d_in, const int* in_sizes, int n_in,
                              void* d_out, int out_size, void* d_ws, size_t ws_size,
                              hipStream_t stream)
{
    (void)in_sizes; (void)n_in; (void)out_size; (void)ws_size;
    const float* x       = (const float*)d_in[0];
    const float* w_embed = (const float*)d_in[1];
    const float* b_embed = (const float*)d_in[2];
    const float* w_out   = (const float*)d_in[3];
    const float* b_out   = (const float*)d_in[4];
    float* out = (float*)d_out;

    _Float16* qws = (_Float16*)d_ws;
    _Float16* kws = qws + QKV_ELEMS;
    _Float16* vws = kws + QKV_ELEMS;
    _Float16* yws = vws + QKV_ELEMS;

    embed_kernel<<<dim3(NTOK / 64, 192 / 4, 2 * NH), 256, 0, stream>>>(
        x, w_embed, b_embed, qws, kws, vws);
    attn_kernel<<<dim3(NTOK / 128, 2 * NH), 128, 0, stream>>>(qws, kws, vws, yws);
    proj_kernel<<<dim3(NTOK / 64, CDIM / 64, 2), 128, 0, stream>>>(
        yws, w_out, b_out, x, out);
}
